// HeteroGATModel_3015067042242
// MI455X (gfx1250) — compile-verified, matrix operand order checked by reading, not run on hardware
//
#include <hip/hip_runtime.h>
#include <math.h>

// ---------------------------------------------------------------------------
// HeteroGAT (2x GATv2 HeteroConv layers) for MI455X / gfx1250 (wave32, WMMA)
// ---------------------------------------------------------------------------

typedef __attribute__((ext_vector_type(16))) _Float16 v16h;
typedef __attribute__((ext_vector_type(8)))  _Float16 v8h;
typedef __attribute__((ext_vector_type(8)))  float    v8f;

#define NEG_SLOPE 0.2f

// ----------------------------- fill ----------------------------------------
__global__ void k_fill(float* __restrict__ p, float v, long n) {
  long i = (long)blockIdx.x * blockDim.x + threadIdx.x;
  if (i < n) p[i] = v;
}

// -------- pack X: f32 [N,K] -> f16 [N,KPAD], zero-padded in K ---------------
__global__ void k_pack_x(const float* __restrict__ X, _Float16* __restrict__ Xh,
                         int N, int K, int KPAD) {
  long t = (long)blockIdx.x * blockDim.x + threadIdx.x;
  if (t >= (long)N * KPAD) return;
  int n = (int)(t / KPAD), k = (int)(t % KPAD);
  Xh[t] = (k < K) ? (_Float16)X[(size_t)n * K + k] : (_Float16)0.f;
}

// ---- pack W^T: f32 W[K,M] -> f16 Wt[M,KPAD] (transposed, zero-padded) ------
__global__ void k_pack_wt(const float* __restrict__ W, _Float16* __restrict__ Wt,
                          int K, int M, int KPAD) {
  int t = (int)((long)blockIdx.x * blockDim.x + threadIdx.x);
  if (t >= M * KPAD) return;
  int m = t / KPAD, k = t % KPAD;
  Wt[t] = (k < K) ? (_Float16)W[(size_t)k * M + m] : (_Float16)0.f;
}

// ------------------- GEMM: Y[N,M] = Xh[N,KPAD] @ Wt[M,KPAD]^T + b -----------
// One wave32 computes a full 16-row x M stripe: A loaded once per K-step
// (two 16B b128 loads), reused across all MT column tiles. Compile-time
// KPAD/MT -> fully unrolled, branch-free, EXEC all-1s across every WMMA.
// A-lane layout per CDNA5 ISA 7.12.2 (16-bit A 16x32): lane half h selects
// K chunks [k0+half*8, +8) and [k0+16+half*8, +8). B mirrored with N=lane&15.
template <int KPAD, int MT, typename OutT>
__global__ void k_gemm_wmma(const _Float16* __restrict__ Xh,
                            const _Float16* __restrict__ Wt,
                            const float* __restrict__ bias, OutT* __restrict__ Y,
                            int N) {
  constexpr int M = MT * 16;
  int wave = (int)(((long)blockIdx.x * blockDim.x + threadIdx.x) >> 5);
  int lane = threadIdx.x & 31;
  int tn = wave;                       // 16-row stripe index
  if (tn * 16 >= N) return;            // wave-uniform exit
  int half = lane >> 4;
  int l16  = lane & 15;
  int rowA = tn * 16 + l16;
  if (rowA >= N) rowA = N - 1;         // clamp loads; stores guarded

  v8f acc[MT] = {};
  const _Float16* xrow = Xh + (size_t)rowA * KPAD;
#pragma unroll
  for (int k0 = 0; k0 < KPAD; k0 += 32) {
    v8h a0 = *(const v8h*)(xrow + k0 + (half << 3));
    v8h a1 = *(const v8h*)(xrow + k0 + 16 + (half << 3));
    v16h a;
#pragma unroll
    for (int i = 0; i < 8; ++i) { a[i] = a0[i]; a[8 + i] = a1[i]; }
#pragma unroll
    for (int tm = 0; tm < MT; ++tm) {
      const _Float16* wrow = Wt + (size_t)(tm * 16 + l16) * KPAD;
      v8h b0 = *(const v8h*)(wrow + k0 + (half << 3));
      v8h b1 = *(const v8h*)(wrow + k0 + 16 + (half << 3));
      v16h b;
#pragma unroll
      for (int i = 0; i < 8; ++i) { b[i] = b0[i]; b[8 + i] = b1[i]; }
      // (neg_a, A, neg_b, B, c_mod, C, reuse_a, reuse_b)
      acc[tm] = __builtin_amdgcn_wmma_f32_16x16x32_f16(false, a, false, b,
                                                       (short)0, acc[tm],
                                                       false, false);
    }
  }
#pragma unroll
  for (int tm = 0; tm < MT; ++tm) {
#pragma unroll
    for (int r = 0; r < 8; ++r) {
      int orow = tn * 16 + r + (half << 3);  // D: VGPR r -> M=r (+8 hi half)
      int ocol = tm * 16 + l16;
      if (orow < N) Y[(size_t)orow * M + ocol] = (OutT)(acc[tm][r] + bias[ocol]);
    }
  }
}

// --------------------- float atomic max (sign-split trick) -------------------
__device__ inline void atomic_max_f32(float* addr, float v) {
  if (v >= 0.0f) atomicMax((int*)addr, __float_as_int(v));
  else           atomicMin((unsigned int*)addr, __float_as_uint(v));
}

// --------------------- pass A: per-edge logits + segment max -----------------
// One wave per edge; lane handles 4 channels (float4). Lanes 0-15 = head 0
// (c 0..63), lanes 16-31 = head 1 (c 64..127). att flat [2,64] == [128].
__global__ void k_edge_logits(const int* __restrict__ ei, const float* __restrict__ ea,
                              const float* __restrict__ We, const float* __restrict__ att,
                              const float* __restrict__ xl, const float* __restrict__ xr,
                              float* __restrict__ logit, float* __restrict__ smax, int E) {
  int wid = (int)(((long)blockIdx.x * blockDim.x + threadIdx.x) >> 5);
  if (wid >= E) return;                // wave-uniform
  int lane = threadIdx.x & 31;
  int src = ei[wid];
  int dst = ei[E + wid];
  float a0 = ea[2 * wid], a1 = ea[2 * wid + 1];
  int c = lane << 2;
  const float4 vl = *(const float4*)(xl + (size_t)src * 128 + c);
  const float4 vr = *(const float4*)(xr + (size_t)dst * 128 + c);
  const float4 w0 = *(const float4*)(We + c);
  const float4 w1 = *(const float4*)(We + 128 + c);
  const float4 at = *(const float4*)(att + c);
  float m0 = vl.x + vr.x + a0 * w0.x + a1 * w1.x;
  float m1 = vl.y + vr.y + a0 * w0.y + a1 * w1.y;
  float m2 = vl.z + vr.z + a0 * w0.z + a1 * w1.z;
  float m3 = vl.w + vr.w + a0 * w0.w + a1 * w1.w;
  m0 = m0 > 0.f ? m0 : NEG_SLOPE * m0;
  m1 = m1 > 0.f ? m1 : NEG_SLOPE * m1;
  m2 = m2 > 0.f ? m2 : NEG_SLOPE * m2;
  m3 = m3 > 0.f ? m3 : NEG_SLOPE * m3;
  float part = m0 * at.x + m1 * at.y + m2 * at.z + m3 * at.w;
  part += __shfl_xor(part, 1);         // reduce within 16-lane half (one head)
  part += __shfl_xor(part, 2);
  part += __shfl_xor(part, 4);
  part += __shfl_xor(part, 8);
  if ((lane & 15) == 0) {
    int h = lane >> 4;
    logit[2 * wid + h] = part;
    atomic_max_f32(&smax[2 * (size_t)dst + h], part);
  }
}

// ------------------ pass B: exp(logit - max) + segment sum -------------------
__global__ void k_edge_exp(const int* __restrict__ ei, const float* __restrict__ smax,
                           float* __restrict__ logit, float* __restrict__ ssum, int E) {
  long t = (long)blockIdx.x * blockDim.x + threadIdx.x;
  if (t >= (long)E * 2) return;
  int e = (int)(t >> 1), h = (int)(t & 1);
  int dst = ei[E + e];
  float mx = smax[2 * (size_t)dst + h];
  if (mx < -3.0e38f) mx = 0.0f;        // empty-segment guard (matches isfinite)
  float ex = __expf(logit[t] - mx);
  logit[t] = ex;                       // overwrite logit with numerator
  atomicAdd(&ssum[2 * (size_t)dst + h], ex);
}

// ---------------- pass C: alpha-weighted scatter-add of messages -------------
__global__ void k_edge_scatter(const int* __restrict__ ei, const float* __restrict__ ex,
                               const float* __restrict__ ssum, const float* __restrict__ xl,
                               float* __restrict__ acc, int E) {
  int wid = (int)(((long)blockIdx.x * blockDim.x + threadIdx.x) >> 5);
  if (wid >= E) return;
  int lane = threadIdx.x & 31;
  int src = ei[wid];
  int dst = ei[E + wid];
  int c = lane << 2;
  int h = c >> 6;
  float alpha = ex[2 * wid + h] / (ssum[2 * (size_t)dst + h] + 1e-16f);
  const float4 vl = *(const float4*)(xl + (size_t)src * 128 + c);
  float* p = acc + (size_t)dst * 128 + c;
  atomicAdd(p + 0, vl.x * alpha);
  atomicAdd(p + 1, vl.y * alpha);
  atomicAdd(p + 2, vl.z * alpha);
  atomicAdd(p + 3, vl.w * alpha);
}

// --------- finalize: mean over heads + conv biases + ReLU -> f16 features ----
__global__ void k_finalize(const float* __restrict__ acc, const float* __restrict__ b1,
                           const float* __restrict__ b2, _Float16* __restrict__ out,
                           int N) {
  long t = (long)blockIdx.x * blockDim.x + threadIdx.x;
  if (t >= (long)N * 64) return;
  int n = (int)(t >> 6), c = (int)(t & 63);
  float v = 0.5f * (acc[(size_t)n * 128 + c] + acc[(size_t)n * 128 + 64 + c]) + b1[c];
  if (b2) v += b2[c];
  out[t] = (_Float16)(v > 0.f ? v : 0.f);
}

// ------------------------- output projection [64 -> 1] ----------------------
__global__ void k_outproj(const _Float16* __restrict__ x, const float* __restrict__ W,
                          const float* __restrict__ b, float* __restrict__ out, int N) {
  int n = (int)((long)blockIdx.x * blockDim.x + threadIdx.x);
  if (n >= N) return;
  const _Float16* r = x + (size_t)n * 64;
  float s = b[0];
#pragma unroll
  for (int k = 0; k < 64; ++k) s += (float)r[k] * W[k];
  out[n] = s;
}

// ---------------------------------------------------------------------------
extern "C" void kernel_launch(void* const* d_in, const int* in_sizes, int n_in,
                              void* d_out, int out_size, void* d_ws, size_t ws_size,
                              hipStream_t stream) {
  const float* x_project = (const float*)d_in[0];
  const float* x_org     = (const float*)d_in[1];
  const int*   ei_of = (const int*)d_in[2];
  const int*   ei_pp = (const int*)d_in[3];
  const int*   ei_fo = (const int*)d_in[4];
  const float* ea_of = (const float*)d_in[5];
  const float* ea_pp = (const float*)d_in[6];
  const float* ea_fo = (const float*)d_in[7];
  const int NP = in_sizes[0] / 32;
  const int NO = in_sizes[1] / 16;
  const int E  = in_sizes[2] / 2;

  // params flattened in setup_inputs() dict insertion order (DFS):
  //  8:lin_P.W  9:lin_P.b  10:lin_O.W  11:lin_O.b
  //  per conv (7 leaves): Wl, bl, Wr, br, We, att, bias
  //  12..18 c1.of | 19..25 c1.pp | 26..32 c1.fo
  //  33..39 c2.of | 40..46 c2.pp | 47..53 c2.fo
  //  54:out.W  55:out.b
  const float* P[64];
  for (int i = 0; i < n_in && i < 64; ++i) P[i] = (const float*)d_in[i];

  // ---- workspace carve-up (bytes, 64B aligned) ----
  char* ws = (char*)d_ws;
  size_t off = 0;
  auto take = [&](size_t bytes) {
    char* p = ws + off; off += (bytes + 63) & ~(size_t)63; return (void*)p;
  };
  _Float16* xpA   = (_Float16*)take((size_t)NP * 64 * 2);  // node feats (f16)
  _Float16* xpB   = (_Float16*)take((size_t)NP * 64 * 2);
  _Float16* xoA   = (_Float16*)take((size_t)NO * 64 * 2);
  _Float16* xoB   = (_Float16*)take((size_t)NO * 64 * 2);
  _Float16* xPh   = (_Float16*)take((size_t)NP * 32 * 2);  // packed inputs
  _Float16* xOh   = (_Float16*)take((size_t)NO * 32 * 2);  // (K=16 padded to 32)
  _Float16* wt1   = (_Float16*)take((size_t)128 * 64 * 2); // packed W^T
  _Float16* wt2   = (_Float16*)take((size_t)128 * 64 * 2);
  float*    xlb   = (float*)take((size_t)NP * 128 * 4);    // xl (f32, edge phase)
  float*    xrb   = (float*)take((size_t)NP * 128 * 4);    // xr
  float*    accP  = (float*)take((size_t)NP * 128 * 4);    // per-head acc, dst=P
  float*    accO  = (float*)take((size_t)NO * 128 * 4);    // per-head acc, dst=O
  float*    lg    = (float*)take((size_t)E * 2 * 4);       // logits -> exp
  float*    smax  = (float*)take((size_t)NP * 2 * 4);
  float*    ssum  = (float*)take((size_t)NP * 2 * 4);
  (void)ws_size;

  auto fill = [&](float* p, float v, long n) {
    long g = (n + 255) / 256;
    k_fill<<<dim3((unsigned)g), dim3(256), 0, stream>>>(p, v, n);
  };
  auto packw = [&](const float* Wf, _Float16* wt, int K, int M, int KPAD) {
    int tot = M * KPAD;
    k_pack_wt<<<dim3((tot + 255) / 256), dim3(256), 0, stream>>>(Wf, wt, K, M, KPAD);
  };
  // conv projection: Y[N,128] f32 = Xh[N,64] @ W[64,128] + b
  auto gemm_conv = [&](const _Float16* Xh, const float* Wf, const float* b,
                       float* Y, int N, _Float16* wtbuf) {
    packw(Wf, wtbuf, 64, 128, 64);
    long g = (((long)((N + 15) >> 4)) * 32 + 255) / 256;
    k_gemm_wmma<64, 8, float><<<dim3((unsigned)g), dim3(256), 0, stream>>>(
        Xh, wtbuf, b, Y, N);
  };
  // input projection: Y[N,64] f16 = Xh[N,32] @ W[K,64] + b  (K=32 or 16)
  auto gemm_lin = [&](const _Float16* Xh, const float* Wf, const float* b,
                      _Float16* Y, int N, int K, _Float16* wtbuf) {
    packw(Wf, wtbuf, K, 64, 32);
    long g = (((long)((N + 15) >> 4)) * 32 + 255) / 256;
    k_gemm_wmma<32, 4, _Float16><<<dim3((unsigned)g), dim3(256), 0, stream>>>(
        Xh, wtbuf, b, Y, N);
  };
  auto run_conv = [&](const _Float16* xs, int Ns, const _Float16* xd, int Nd,
                      const int* ei, const float* ea, int pb, float* accDst) {
    gemm_conv(xs, P[pb + 0], P[pb + 1], xlb, Ns, wt1);   // xl = xs@Wl + bl
    gemm_conv(xd, P[pb + 2], P[pb + 3], xrb, Nd, wt2);   // xr = xd@Wr + br
    fill(smax, -INFINITY, (long)Nd * 2);
    fill(ssum, 0.0f, (long)Nd * 2);
    long gw = ((long)E * 32 + 255) / 256;                // wave-per-edge
    long gt = ((long)E * 2 + 255) / 256;
    k_edge_logits<<<dim3((unsigned)gw), dim3(256), 0, stream>>>(
        ei, ea, P[pb + 4], P[pb + 5], xlb, xrb, lg, smax, E);
    k_edge_exp<<<dim3((unsigned)gt), dim3(256), 0, stream>>>(ei, smax, lg, ssum, E);
    k_edge_scatter<<<dim3((unsigned)gw), dim3(256), 0, stream>>>(ei, lg, ssum, xlb,
                                                                 accDst, E);
  };
  auto layer = [&](int base, const _Float16* xpIn, const _Float16* xoIn,
                   _Float16* xpOut, _Float16* xoOut) {
    int of = base, pp = base + 7, fo = base + 14;
    // dst = Project: sum of 'of' and 'pp' convs (HeteroConv aggr='sum')
    fill(accP, 0.0f, (long)NP * 128);
    run_conv(xoIn, NO, xpIn, NP, ei_of, ea_of, of, accP);
    run_conv(xpIn, NP, xpIn, NP, ei_pp, ea_pp, pp, accP);
    { long g = ((long)NP * 64 + 255) / 256;
      k_finalize<<<dim3((unsigned)g), dim3(256), 0, stream>>>(accP, P[of + 6],
                                                              P[pp + 6], xpOut, NP); }
    // dst = Org: 'fo' conv only (uses OLD xp as source -> xpIn still intact)
    fill(accO, 0.0f, (long)NO * 128);
    run_conv(xpIn, NP, xoIn, NO, ei_fo, ea_fo, fo, accO);
    { long g = ((long)NO * 64 + 255) / 256;
      k_finalize<<<dim3((unsigned)g), dim3(256), 0, stream>>>(accO, P[fo + 6],
                                                              nullptr, xoOut, NO); }
  };

  // pack raw inputs to f16 (zero-padded K)
  { long n = (long)NP * 32, g = (n + 255) / 256;
    k_pack_x<<<dim3((unsigned)g), dim3(256), 0, stream>>>(x_project, xPh, NP, 32, 32); }
  { long n = (long)NO * 32, g = (n + 255) / 256;
    k_pack_x<<<dim3((unsigned)g), dim3(256), 0, stream>>>(x_org, xOh, NO, 16, 32); }

  // input projections (WMMA)
  gemm_lin(xPh, P[8],  P[9],  xpA, NP, 32, wt1);
  gemm_lin(xOh, P[10], P[11], xoA, NO, 16, wt2);
  // two hetero layers (ping-pong)
  layer(12, xpA, xoA, xpB, xoB);
  layer(33, xpB, xoB, xpA, xoA);
  // final scalar head
  { long g = (NP + 255) / 256;
    k_outproj<<<dim3((unsigned)g), dim3(256), 0, stream>>>(xpA, P[54], P[55],
                                                           (float*)d_out, NP); }
  (void)out_size;
}
